// S4D_50148038148622
// MI455X (gfx1250) — matrix operand order repaired and checked by reading, not compiled, not run on hardware
//
#include <hip/hip_runtime.h>
#include <hip/hip_bf16.h>

// ---------------------------------------------------------------------------
// Problem constants (match reference)
// ---------------------------------------------------------------------------
#define B_ 16
#define H_ 512
#define L_ 2048
#define N2_ 32
#define PI_F 3.14159265358979323846f

#define GLOBAL_AS __attribute__((address_space(1)))
#define LOCAL_AS  __attribute__((address_space(3)))

typedef __attribute__((ext_vector_type(16))) __bf16 vbf16x16;
typedef __attribute__((ext_vector_type(8)))  float  vf32x8;
typedef __attribute__((ext_vector_type(4)))  unsigned int u32x4;
typedef __attribute__((ext_vector_type(8)))  int          i32x8;
typedef __attribute__((ext_vector_type(4)))  int          i32x4;
typedef __attribute__((ext_vector_type(4)))  int          v4i;

#if __has_builtin(__builtin_amdgcn_global_load_async_to_lds_b128)
#define HAVE_ASYNC_LDS 1
#else
#define HAVE_ASYNC_LDS 0
#endif
#if __has_builtin(__builtin_amdgcn_tensor_load_to_lds)
#define HAVE_TDM 1
#else
#define HAVE_TDM 0
#endif

union FragBF {
    vbf16x16     v;
    unsigned int u[8];
};

__device__ __forceinline__ unsigned short f32_to_bf16(float f) {
    union { float f; unsigned int u; } x; x.f = f;
    unsigned int r = x.u + 0x7FFFu + ((x.u >> 16) & 1u);   // round-to-nearest-even
    return (unsigned short)(r >> 16);
}

__device__ __forceinline__ void wait_async0() {
#if __has_builtin(__builtin_amdgcn_s_wait_asynccnt)
    __builtin_amdgcn_s_wait_asynccnt(0);
#else
    asm volatile("s_wait_asynccnt 0x0" ::: "memory");
#endif
}

__device__ __forceinline__ void wait_tensor0() {
#if __has_builtin(__builtin_amdgcn_s_wait_tensorcnt)
    __builtin_amdgcn_s_wait_tensorcnt(0);
#else
    asm volatile("s_wait_tensorcnt 0x0" ::: "memory");
#endif
}

// ---------------------------------------------------------------------------
// Kernel 1: transpose fp32 (R x C) -> bf16 (C x R)
// ---------------------------------------------------------------------------
__global__ __launch_bounds__(256) void transpose_to_bf16(
    const float* __restrict__ src, unsigned short* __restrict__ dst,
    int R, int C) {
    int idx = blockIdx.x * 256 + threadIdx.x;
    if (idx >= R * C) return;
    int r = idx / C, c = idx - r * C;
    dst[(size_t)c * R + r] = f32_to_bf16(src[(size_t)r * C + c]);
}

// ---------------------------------------------------------------------------
// Kernel 2: diagonal-SSM scan (exact causal conv) + D*u + exact GELU -> bf16
// One wave per (b,h) row; lane n = complex mode n (N2 == warpSize == 32).
// ---------------------------------------------------------------------------
__global__ __launch_bounds__(256) void s4d_scan(
    const float* __restrict__ u, const float* __restrict__ Dp,
    const float* __restrict__ log_dt,
    const float* __restrict__ C_re, const float* __restrict__ C_im,
    unsigned short* __restrict__ y0) {

    __shared__ float tile[8][32][33];          // [wave][mode][time], conflict-free
    const int wloc = threadIdx.x >> 5;
    const int lane = threadIdx.x & 31;
    const int gw   = blockIdx.x * 8 + wloc;    // row index b*H + h  (0..8191)
    const int h    = gw & (H_ - 1);

    // z = exp(dt*A), A_n = -1/2 + i*pi*n ; C' = 2*C*(z-1)/A  (factor 2 folded)
    const float dt = expf(log_dt[h]);
    const float ar = -0.5f * dt;
    const float ai = PI_F * (float)lane * dt;
    const float ea = expf(ar);
    const float zr = ea * cosf(ai), zi = ea * sinf(ai);
    const float er = zr - 1.0f,      ei = zi;
    const float Ar = -0.5f,          Ai = PI_F * (float)lane;
    const float inv = 1.0f / (Ar * Ar + Ai * Ai);
    const float wr = (er * Ar + ei * Ai) * inv;
    const float wi = (ei * Ar - er * Ai) * inv;
    const float c0r = C_re[h * N2_ + lane], c0i = C_im[h * N2_ + lane];
    const float Cr = 2.0f * (c0r * wr - c0i * wi);
    const float Ci = 2.0f * (c0r * wi + c0i * wr);
    const float Dh = Dp[h];

    const float* __restrict__ up = u + (size_t)gw * L_;
    unsigned short* __restrict__ yp = y0 + (size_t)gw * L_;

    float xr = 0.0f, xi = 0.0f;
    for (int base = 0; base < L_; base += 32) {
        float uv = up[base + lane];
        #pragma unroll
        for (int t = 0; t < 32; ++t) {
            float ul  = __shfl(uv, t, 32);
            float nxr = zr * xr - zi * xi + ul;
            float nxi = zr * xi + zi * xr;
            xr = nxr; xi = nxi;
            tile[wloc][lane][t] = Cr * xr - Ci * xi;   // Re(C' * x)
        }
        __syncthreads();
        float acc = 0.0f;
        #pragma unroll
        for (int n = 0; n < 32; ++n) acc += tile[wloc][n][lane];
        __syncthreads();
        float v = acc + Dh * uv;
        float g = 0.5f * v * (1.0f + erff(v * 0.70710678f));  // exact GELU
        yp[base + lane] = f32_to_bf16(g);
    }
}

// ---------------------------------------------------------------------------
// Kernel 3: tiled bf16 GEMM  C[M,N] = act(A[M,K] * Bt[N,K]^T + bias)
// Macro tile 128x64, TK=32, 8 waves -> each wave 32x32 via 2x2 WMMA frags.
// A tile fed by GLOBAL_LOAD_ASYNC_TO_LDS_B128 (ASYNCcnt), B tile fed by the
// Tensor Data Mover (TENSORcnt) with LDS row padding done by the TDM.
// ---------------------------------------------------------------------------
#define TM 128
#define TN 64
#define TK 32
#define LDT 40   // padded LDS row (ushorts): 80B, 16B-aligned rows

template<int ACT /*0=none,1=tanh*/, int OUTBF16>
__global__ __launch_bounds__(256) void gemm_bf16(
    const unsigned short* __restrict__ A,   // M x K (bf16 bits)
    const unsigned short* __restrict__ Bt,  // N x K (bf16 bits)
    const float* __restrict__ bias,         // N
    void* __restrict__ Cout,                // M x N
    int M, int N, int K) {

    __shared__ __attribute__((aligned(16))) unsigned short sA[TM][LDT];
    __shared__ __attribute__((aligned(16))) unsigned short sB[TN][LDT];

    const int tid  = threadIdx.x;
    const int lane = tid & 31;
    const int wid  = tid >> 5;       // 0..7
    const int wm   = wid >> 1;       // 0..3 -> M offset wm*32
    const int wn   = wid & 1;        // 0..1 -> N offset wn*32
    const int hi   = lane >> 4;
    const int m0   = blockIdx.y * TM;
    const int n0   = blockIdx.x * TN;

    vf32x8 acc[2][2] = {};

    // per-thread tile-load coordinates
    const int arow = tid >> 1, acb = (tid & 1) << 4;     // A: 128 rows x 32
    const int brow = tid >> 2, bcb = (tid & 3) << 3;     // B:  64 rows x 32
    (void)brow; (void)bcb;

#if HAVE_TDM
    // Uniform pieces of the TDM descriptor (group1 is k0-invariant).
    const unsigned sb_base =
        (unsigned)(unsigned long long)(LOCAL_AS void*)&sB[0][0];
    i32x8 g1;
    // [17:16] data_size=2B; [20] pad_enable; [24:22] pad_interval=16DW;
    // [31:25] pad_amount=4DW  (=> LDS rows of 64B data + 16B pad = LDT)
    g1[0] = (int)0x06D10000u;
    g1[1] = (int)(((unsigned)K & 0xFFFFu) << 16);          // tensor_dim0 lo16
    g1[2] = (int)((((unsigned)K >> 16) & 0xFFFFu) |
                  (((unsigned)N & 0xFFFFu) << 16));        // dim0 hi | dim1 lo
    g1[3] = (int)((((unsigned)N >> 16) & 0xFFFFu) |
                  ((unsigned)TK << 16));                   // dim1 hi | tile_dim0
    g1[4] = (int)(unsigned)TN;                             // tile_dim1 (tile_dim2=0)
    g1[5] = (int)(unsigned)K;                              // tensor_dim0_stride lo32
    g1[6] = 0;                                             // stride hi | dim1_stride lo
    g1[7] = 0;
    const i32x4 gz4 = {0, 0, 0, 0};
    const i32x8 gz8 = {0, 0, 0, 0, 0, 0, 0, 0};
#endif

    for (int k0 = 0; k0 < K; k0 += TK) {
        __syncthreads();   // previous iteration's LDS reads complete

#if HAVE_ASYNC_LDS
        {   // A tile -> LDS via async DMA (2 x b128 per thread)
            GLOBAL_AS v4i* src = (GLOBAL_AS v4i*)(const void*)
                (A + (size_t)(m0 + arow) * K + k0 + acb);
            LOCAL_AS v4i* dst = (LOCAL_AS v4i*)&sA[arow][acb];
            __builtin_amdgcn_global_load_async_to_lds_b128(src, dst, 0, 0);
            __builtin_amdgcn_global_load_async_to_lds_b128(src, dst, 16, 0);
        }
#else
        {   // A tile -> LDS (2 x b128 per thread, through VGPRs)
            const uint4* s = reinterpret_cast<const uint4*>(
                A + (size_t)(m0 + arow) * K + k0 + acb);
            uint4* d = reinterpret_cast<uint4*>(&sA[arow][acb]);
            d[0] = s[0]; d[1] = s[1];
        }
#endif

#if HAVE_TDM
        if (wid == 0) {   // one TDM op per workgroup per K-step (EXEC-independent)
            unsigned long long ga =
                (unsigned long long)(Bt + (size_t)n0 * K + k0);
            u32x4 g0;
            g0[0] = 1u;                                    // count=1, user mode
            g0[1] = sb_base;                               // lds_addr
            g0[2] = (unsigned)(ga & 0xFFFFFFFFu);          // global_addr lo
            g0[3] = (unsigned)((ga >> 32) & 0x01FFFFFFu) | (2u << 30); // hi|type=2
            __builtin_amdgcn_tensor_load_to_lds(g0, g1, gz4, gz4, gz8, 0);
            wait_tensor0();
        }
#else
        {   // B tile -> LDS (1 x b128 per thread, through VGPRs)
            const uint4* s = reinterpret_cast<const uint4*>(
                Bt + (size_t)(n0 + brow) * K + k0 + bcb);
            uint4* d = reinterpret_cast<uint4*>(&sB[brow][bcb]);
            d[0] = s[0];
        }
#endif

        if (k0 + TK < K)   // warm L2 for the next K-slab
            __builtin_prefetch(A + (size_t)(m0 + arow) * K + k0 + TK + acb, 0, 1);

#if HAVE_ASYNC_LDS
        wait_async0();
#endif
        __syncthreads();

        // Fragments per documented 16-bit WMMA VGPR layouts
        FragBF afrag[2], bfrag[2];
        #pragma unroll
        for (int mi = 0; mi < 2; ++mi) {
            const unsigned short* p = &sA[wm * 32 + mi * 16 + (lane & 15)][0];
            #pragma unroll
            for (int j = 0; j < 8; ++j) {
                int k = ((j >> 2) << 4) + (hi << 3) + ((j & 3) << 1);
                afrag[mi].u[j] = *reinterpret_cast<const unsigned int*>(p + k);
            }
        }
        #pragma unroll
        for (int ni = 0; ni < 2; ++ni) {
            const unsigned short* p = &sB[wn * 32 + ni * 16 + (lane & 15)][0];
            #pragma unroll
            for (int j = 0; j < 8; ++j) {
                int k = ((j >> 2) << 4) + (hi << 3) + ((j & 3) << 1);
                bfrag[ni].u[j] = *reinterpret_cast<const unsigned int*>(p + k);
            }
        }
        #pragma unroll
        for (int mi = 0; mi < 2; ++mi)
            #pragma unroll
            for (int ni = 0; ni < 2; ++ni)
                acc[mi][ni] = __builtin_amdgcn_wmma_f32_16x16x32_bf16(
                    false, afrag[mi].v, false, bfrag[ni].v,
                    (short)0, acc[mi][ni], false, false);
    }

    // Epilogue: C/D layout (N = lane&15, M = hi*8 + r), fused bias + activation
    float* of          = (float*)Cout;
    unsigned short* ob = (unsigned short*)Cout;
    #pragma unroll
    for (int mi = 0; mi < 2; ++mi) {
        #pragma unroll
        for (int ni = 0; ni < 2; ++ni) {
            int col = n0 + wn * 32 + ni * 16 + (lane & 15);
            float bv = bias[col];
            #pragma unroll
            for (int r = 0; r < 8; ++r) {
                int row = m0 + wm * 32 + mi * 16 + hi * 8 + r;
                float v = acc[mi][ni][r] + bv;
                if (ACT == 1) v = tanhf(v);
                if (OUTBF16) ob[(size_t)row * N + col] = f32_to_bf16(v);
                else         of[(size_t)row * N + col] = v;
            }
        }
    }
}

// ---------------------------------------------------------------------------
// Launcher
// ---------------------------------------------------------------------------
extern "C" void kernel_launch(void* const* d_in, const int* in_sizes, int n_in,
                              void* d_out, int out_size, void* d_ws, size_t ws_size,
                              hipStream_t stream) {
    const float* u     = (const float*)d_in[0];
    const float* Dv    = (const float*)d_in[1];
    const float* logdt = (const float*)d_in[2];
    const float* Cre   = (const float*)d_in[3];
    const float* Cim   = (const float*)d_in[4];
    const float* W1    = (const float*)d_in[5];
    const float* b1    = (const float*)d_in[6];
    const float* W2    = (const float*)d_in[7];
    const float* b2    = (const float*)d_in[8];
    float* out = (float*)d_out;

    const int M  = B_ * H_;      // 8192
    const int N1 = 2 * H_;       // 1024
    const int N2o = H_;          // 512
    const int K1 = L_;           // 2048

    // workspace layout (bf16 bits)
    unsigned short* y0bf = (unsigned short*)d_ws;             // M  x L
    unsigned short* w1t  = y0bf + (size_t)M * L_;             // N1 x L
    unsigned short* act1 = w1t  + (size_t)N1 * L_;            // M  x N1
    unsigned short* w2t  = act1 + (size_t)M * N1;             // N2o x N1

    // 1) weight transpose + bf16 convert
    transpose_to_bf16<<<(L_ * N1 + 255) / 256, 256, 0, stream>>>(W1, w1t, L_, N1);
    transpose_to_bf16<<<(N1 * N2o + 255) / 256, 256, 0, stream>>>(W2, w2t, N1, N2o);

    // 2) diagonal-SSM scan + D*u + GELU  (exact causal convolution)
    s4d_scan<<<(B_ * H_) / 8, 256, 0, stream>>>(u, Dv, logdt, Cre, Cim, y0bf);

    // 3) GEMM1: (M,K1) x (K1,N1) + b1, tanh, bf16 out
    gemm_bf16<1, 1><<<dim3(N1 / TN, M / TM), 256, 0, stream>>>(
        y0bf, w1t, b1, (void*)act1, M, N1, K1);

    // 4) GEMM2: (M,N1) x (N1,N2o) + b2, f32 out
    gemm_bf16<0, 0><<<dim3(N2o / TN, M / TM), 256, 0, stream>>>(
        act1, w2t, b2, (void*)out, M, N2o, N1);
}
